// CausalMultiHeadSelfAttention_85040352461191
// MI455X (gfx1250) — matrix operand order repaired and checked by reading, not compiled, not run on hardware
//
#include <hip/hip_runtime.h>

// ---------------------------------------------------------------------------
// Causal multi-head self-attention, MI455X (gfx1250), wave32 + WMMA f16.
//  Pass 0: convert x, Wq, Wk, Wv, Wo to f16 (one-time, bandwidth-trivial).
//  Pass 1: Q = x Wq^T ; K = x Wk^T  (f16, row-major [8192,512])
//          V = x Wv^T + bv          (f16, TRANSPOSED per head: Vt[b][h][d][s])
//  Pass 2: flash attention, causal, f32 online softmax in exp2 domain;
//          mask-free fast path for sub-diagonal blocks; waves independent
//          (no __syncthreads; only per-wave LDS for P layout round trip).
//  Pass 3: y = ctx Wo^T + bo (f32 output).
// All matmuls via v_wmma_f32_16x16x32_f16 (f32 accumulate).
// ---------------------------------------------------------------------------

typedef __attribute__((ext_vector_type(16))) _Float16 v16h;
typedef __attribute__((ext_vector_type(8)))  _Float16 v8h;
typedef __attribute__((ext_vector_type(8)))  float    v8f;

#define D_MODEL 512
#define SEQ     4096
#define NHEADS  8
#define DK      64
#define NBATCH  2
#define MROWS   (NBATCH * SEQ)   // 8192
#define QBLOCKS (SEQ / 64)       // 64

// 1/sqrt(d_k) * log2(e): softmax runs in exp2 domain (v_exp_f32 is exp2).
#define SCALE_LOG2 0.18033688011112042f
#define NEG_BIG    -3.0e38f

__device__ __forceinline__ v8f wmma_f16(const v16h& a, const v16h& b, const v8f& c) {
  return __builtin_amdgcn_wmma_f32_16x16x32_f16(
      /*neg_a=*/false, a, /*neg_b=*/false, b,
      /*c_mod=*/(short)0, c, /*reuse_a=*/false, /*reuse_b=*/false);
}

// A fragment (16x32, f16): lane m = lane%16; halves 0..7 <- k0+8g+i,
// halves 8..15 <- k0+16+8g+i.  `row` points at k=0 of this lane's row.
__device__ __forceinline__ v16h load_a_f16(const _Float16* __restrict__ row, int g, int k0) {
  v8h lo = *(const v8h*)(row + k0 + 8 * g);
  v8h hi = *(const v8h*)(row + k0 + 16 + 8 * g);
  v16h a;
#pragma unroll
  for (int i = 0; i < 8; ++i) { a[i] = lo[i]; a[8 + i] = hi[i]; }
  return a;
}

// B fragment (32x16, f16): lane n = lane%16; halves 0..7 <- k0+8g+i,
// halves 8..15 <- k0+16+8g+i.
__device__ __forceinline__ v16h load_b_f16(const _Float16* __restrict__ row, int g, int k0) {
  v8h lo = *(const v8h*)(row + k0 + 8 * g);
  v8h hi = *(const v8h*)(row + k0 + 16 + 8 * g);
  v16h b;
#pragma unroll
  for (int i = 0; i < 8; ++i) { b[i] = lo[i]; b[8 + i] = hi[i]; }
  return b;
}

// ---------------------------------------------------------------------------
// One-time fp32 -> f16 conversion (n must be a multiple of 8).
// ---------------------------------------------------------------------------
__global__ __launch_bounds__(256) void cvt_f16_kernel(
    const float* __restrict__ src, _Float16* __restrict__ dst, int n) {
  const int i = (blockIdx.x * 256 + threadIdx.x) * 8;
  if (i < n) {
    const float4 a = *(const float4*)(src + i);
    const float4 b = *(const float4*)(src + i + 4);
    v8h o;
    o[0] = (_Float16)a.x; o[1] = (_Float16)a.y;
    o[2] = (_Float16)a.z; o[3] = (_Float16)a.w;
    o[4] = (_Float16)b.x; o[5] = (_Float16)b.y;
    o[6] = (_Float16)b.z; o[7] = (_Float16)b.w;
    *(v8h*)(dst + i) = o;
  }
}

// ---------------------------------------------------------------------------
// Projection: acc[m,n] = sum_k Xh[m,k] * Wh[n,k]  (all f16 in, f32 accum)
// Block = 128 threads (4 waves); block tile 64x64; wave tile 16x64.
// vtrans == 0 : Y[row*512 + col] = acc (+0)       (row-major, Q/K)
// vtrans == 1 : Y = Vt[b][h][d][s] layout          (head-transposed, V)
// ---------------------------------------------------------------------------
__global__ __launch_bounds__(128) void qkv_proj_kernel(
    const _Float16* __restrict__ Xh, const _Float16* __restrict__ Wh,
    const float* __restrict__ bias, _Float16* __restrict__ Y, int vtrans) {
  const int lane = threadIdx.x & 31;
  const int wv   = threadIdx.x >> 5;
  const int g    = lane >> 4;
  const int nl   = lane & 15;
  const int rowBase = blockIdx.x * 64 + wv * 16;
  const int colBase = blockIdx.y * 64;

  const _Float16* arow = Xh + (size_t)(rowBase + nl) * D_MODEL;

  v8f acc[4];
#pragma unroll
  for (int t = 0; t < 4; ++t)
#pragma unroll
    for (int i = 0; i < 8; ++i) acc[t][i] = 0.f;

  for (int k0 = 0; k0 < D_MODEL; k0 += 32) {
    v16h a = load_a_f16(arow, g, k0);
#pragma unroll
    for (int t = 0; t < 4; ++t) {
      const _Float16* wrow = Wh + (size_t)(colBase + t * 16 + nl) * D_MODEL;
      v16h b = load_b_f16(wrow, g, k0);
      acc[t] = wmma_f16(a, b, acc[t]);
    }
  }

  if (vtrans) {
#pragma unroll
    for (int t = 0; t < 4; ++t) {
      const int col = colBase + t * 16 + nl;       // = h*64 + d
      const float bval = bias[col];
      const int h = col >> 6, d = col & 63;
#pragma unroll
      for (int v = 0; v < 8; ++v) {
        const int row = rowBase + v + 8 * g;       // = b*4096 + s
        const int bb = row >> 12, s = row & 4095;
        Y[(((size_t)bb * NHEADS + h) * DK + d) * SEQ + s] =
            (_Float16)(acc[t][v] + bval);
      }
    }
  } else {
#pragma unroll
    for (int t = 0; t < 4; ++t) {
      const int col = colBase + t * 16 + nl;
#pragma unroll
      for (int v = 0; v < 8; ++v) {
        const int row = rowBase + v + 8 * g;
        Y[(size_t)row * D_MODEL + col] = (_Float16)acc[t][v];
      }
    }
  }
}

// ---------------------------------------------------------------------------
// Output projection: Y_f32[m,n] = sum_k Cx[m,k] * Woh[n,k] + bias[n]
// ---------------------------------------------------------------------------
__global__ __launch_bounds__(128) void out_proj_kernel(
    const _Float16* __restrict__ Cx, const _Float16* __restrict__ Wh,
    const float* __restrict__ bias, float* __restrict__ Y) {
  const int lane = threadIdx.x & 31;
  const int wv   = threadIdx.x >> 5;
  const int g    = lane >> 4;
  const int nl   = lane & 15;
  const int rowBase = blockIdx.x * 64 + wv * 16;
  const int colBase = blockIdx.y * 64;

  const _Float16* arow = Cx + (size_t)(rowBase + nl) * D_MODEL;

  v8f acc[4];
#pragma unroll
  for (int t = 0; t < 4; ++t)
#pragma unroll
    for (int i = 0; i < 8; ++i) acc[t][i] = 0.f;

  for (int k0 = 0; k0 < D_MODEL; k0 += 32) {
    v16h a = load_a_f16(arow, g, k0);
#pragma unroll
    for (int t = 0; t < 4; ++t) {
      const _Float16* wrow = Wh + (size_t)(colBase + t * 16 + nl) * D_MODEL;
      v16h b = load_b_f16(wrow, g, k0);
      acc[t] = wmma_f16(a, b, acc[t]);
    }
  }

#pragma unroll
  for (int t = 0; t < 4; ++t) {
    const int col = colBase + t * 16 + nl;
    const float bval = bias[col];
#pragma unroll
    for (int v = 0; v < 8; ++v) {
      const int row = rowBase + v + 8 * g;
      Y[(size_t)row * D_MODEL + col] = acc[t][v] + bval;
    }
  }
}

// ---------------------------------------------------------------------------
// One key-block step of flash attention for one wave (16 q-rows x 64 keys).
// MASKED = true only for the diagonal block; the branch is uniform, so EXEC
// remains all-ones around every WMMA.
// ---------------------------------------------------------------------------
template <bool MASKED>
__device__ __forceinline__ void attn_step(
    int kb, int qBase, int g, int nl,
    const _Float16* __restrict__ kbase, const _Float16* __restrict__ vbase,
    const v16h& qa0, const v16h& qa1,
    v8f (&oacc)[4], float (&mrow)[8], float (&lrow)[8],
    _Float16 (&Pb)[16][72]) {
  // S = Q K^T (4 tiles of 16x16, 2 K=32 steps each).
  v8f sacc[4];
#pragma unroll
  for (int t = 0; t < 4; ++t)
#pragma unroll
    for (int i = 0; i < 8; ++i) sacc[t][i] = 0.f;

#pragma unroll
  for (int t = 0; t < 4; ++t) {
    const _Float16* krow = kbase + (size_t)(kb * 64 + t * 16 + nl) * D_MODEL;
    v16h b0 = load_b_f16(krow, g, 0);
    sacc[t] = wmma_f16(qa0, b0, sacc[t]);
    v16h b1 = load_b_f16(krow, g, 32);
    sacc[t] = wmma_f16(qa1, b1, sacc[t]);
  }

  // Online softmax update in exp2 domain.
#pragma unroll
  for (int v = 0; v < 8; ++v) {
    float rm = mrow[v];
#pragma unroll
    for (int t = 0; t < 4; ++t) {
      float s = sacc[t][v] * SCALE_LOG2;
      if (MASKED) {
        const int q   = qBase + v + 8 * g;
        const int key = kb * 64 + t * 16 + nl;
        if (key > q) s = NEG_BIG;                // causal mask (diagonal only)
      }
      sacc[t][v] = s;
      rm = fmaxf(rm, s);
    }
#pragma unroll
    for (int off = 1; off < 16; off <<= 1)
      rm = fmaxf(rm, __shfl_xor(rm, off, 16));

    const float corr = exp2f(mrow[v] - rm);      // underflows to 0 on first hit
    lrow[v] *= corr;
#pragma unroll
    for (int t = 0; t < 4; ++t) oacc[t][v] *= corr;

    float ps = 0.f;
    const int r = v + 8 * g;                     // local row in C layout
#pragma unroll
    for (int t = 0; t < 4; ++t) {
      const float p = exp2f(sacc[t][v] - rm);
      ps += p;
      Pb[r][t * 16 + nl] = (_Float16)p;
    }
#pragma unroll
    for (int off = 1; off < 16; off <<= 1) ps += __shfl_xor(ps, off, 16);
    lrow[v] += ps;
    mrow[v] = rm;
  }

  // O += P x V  (A frags from per-wave LDS; B frags contiguous from Vt).
  // Same-wave DS ordering guarantees Pb stores land before these loads.
  v16h pa0 = load_a_f16(&Pb[nl][0], g, 0);
  v16h pa1 = load_a_f16(&Pb[nl][0], g, 32);
#pragma unroll
  for (int t = 0; t < 4; ++t) {
    const _Float16* vrow = vbase + (size_t)(t * 16 + nl) * SEQ + kb * 64;
    v16h vb0 = load_b_f16(vrow, g, 0);
    oacc[t] = wmma_f16(pa0, vb0, oacc[t]);
    v16h vb1 = load_b_f16(vrow, g, 32);
    oacc[t] = wmma_f16(pa1, vb1, oacc[t]);
  }
}

// ---------------------------------------------------------------------------
// Flash attention, causal. Grid: (QBLOCKS, NBATCH*NHEADS). Block = 128 (4 waves).
// Long (high-qBlock) workgroups launch first for better tail packing.
// ---------------------------------------------------------------------------
__global__ __launch_bounds__(128) void attn_kernel(
    const _Float16* __restrict__ Qh, const _Float16* __restrict__ Kh,
    const _Float16* __restrict__ Vt, _Float16* __restrict__ Ctx) {
  // Per-wave P tile (16 q-rows x 64 keys): C-layout -> A-layout round trip.
  // Rows padded to 72 halves (144 B) to keep 16 B alignment of all chunks.
  __shared__ alignas(16) _Float16 Pbuf[4][16][72];

  const int tid  = threadIdx.x;
  const int lane = tid & 31;
  const int wv   = tid >> 5;
  const int g    = lane >> 4;
  const int nl   = lane & 15;

  const int bh = blockIdx.y;
  const int bb = bh >> 3;          // batch
  const int h  = bh & 7;           // head
  const _Float16* kbase = Kh + (size_t)bb * SEQ * D_MODEL + (size_t)h * DK;
  const _Float16* vbase = Vt + ((size_t)bh * DK) * SEQ;   // + d*SEQ + s

  const int qBlock = (QBLOCKS - 1) - blockIdx.x;   // longest work first
  const int qBase  = qBlock * 64 + wv * 16;

  // Q fragments for this wave's 16 rows (d_k = 64 -> two K=32 fragments).
  const _Float16* qrow = Qh + (size_t)bb * SEQ * D_MODEL + (size_t)h * DK +
                         (size_t)(qBase + nl) * D_MODEL;
  const v16h qa0 = load_a_f16(qrow, g, 0);
  const v16h qa1 = load_a_f16(qrow, g, 32);

  v8f oacc[4];
#pragma unroll
  for (int t = 0; t < 4; ++t)
#pragma unroll
    for (int i = 0; i < 8; ++i) oacc[t][i] = 0.f;

  float mrow[8], lrow[8];
#pragma unroll
  for (int v = 0; v < 8; ++v) { mrow[v] = NEG_BIG; lrow[v] = 0.f; }

  // Mask-free sweep over fully-visible key blocks.
  for (int kb = 0; kb < qBlock; ++kb) {
    // Prefetch next key block (K rows + Vt rows). -> global_prefetch_b8
    const int nb = (kb + 1) * 64;
    if (tid < 64)
      __builtin_prefetch(kbase + (size_t)(nb + tid) * D_MODEL, 0, 1);
    else
      __builtin_prefetch(vbase + (size_t)(tid - 64) * SEQ + nb, 0, 1);

    attn_step<false>(kb, qBase, g, nl, kbase, vbase, qa0, qa1,
                     oacc, mrow, lrow, Pbuf[wv]);
  }
  // Diagonal block with causal mask.
  attn_step<true>(qBlock, qBase, g, nl, kbase, vbase, qa0, qa1,
                  oacc, mrow, lrow, Pbuf[wv]);

  // Normalize and store context (f16) back into [8192, 512] layout.
#pragma unroll
  for (int v = 0; v < 8; ++v) {
    const float inv = 1.0f / lrow[v];
    const int row = qBase + v + 8 * g;
#pragma unroll
    for (int t = 0; t < 4; ++t) {
      Ctx[((size_t)bb * SEQ + row) * D_MODEL + (size_t)h * DK + t * 16 + nl] =
          (_Float16)(oacc[t][v] * inv);
    }
  }
}

// ---------------------------------------------------------------------------
extern "C" void kernel_launch(void* const* d_in, const int* in_sizes, int n_in,
                              void* d_out, int out_size, void* d_ws, size_t ws_size,
                              hipStream_t stream) {
  (void)in_sizes; (void)n_in; (void)out_size; (void)ws_size;
  const float* x  = (const float*)d_in[0];
  const float* Wq = (const float*)d_in[1];
  const float* Wk = (const float*)d_in[2];
  const float* Wv = (const float*)d_in[3];
  const float* bv = (const float*)d_in[4];
  const float* Wo = (const float*)d_in[5];
  const float* bo = (const float*)d_in[6];
  float* out = (float*)d_out;

  const size_t NX = (size_t)MROWS * D_MODEL;    // 4M elements
  const size_t NW = (size_t)D_MODEL * D_MODEL;  // 256K elements

  _Float16* Xh  = (_Float16*)d_ws;
  _Float16* Wqh = Xh  + NX;
  _Float16* Wkh = Wqh + NW;
  _Float16* Wvh = Wkh + NW;
  _Float16* Woh = Wvh + NW;
  _Float16* Qh  = Woh + NW;
  _Float16* Kh  = Qh  + NX;
  _Float16* Vt  = Kh  + NX;   // [b][h][d][s] head-transposed
  _Float16* Cx  = Vt  + NX;   // total ~42 MB workspace

  // Pass 0: one-time fp32 -> f16 conversions.
  cvt_f16_kernel<<<(int)(NX / 2048), 256, 0, stream>>>(x,  Xh,  (int)NX);
  cvt_f16_kernel<<<(int)(NW / 2048), 256, 0, stream>>>(Wq, Wqh, (int)NW);
  cvt_f16_kernel<<<(int)(NW / 2048), 256, 0, stream>>>(Wk, Wkh, (int)NW);
  cvt_f16_kernel<<<(int)(NW / 2048), 256, 0, stream>>>(Wv, Wvh, (int)NW);
  cvt_f16_kernel<<<(int)(NW / 2048), 256, 0, stream>>>(Wo, Woh, (int)NW);

  // Pass 1: projections (V stored head-transposed with bias).
  dim3 gProj(MROWS / 64, D_MODEL / 64);
  qkv_proj_kernel<<<gProj, 128, 0, stream>>>(Xh, Wqh, nullptr, Qh, 0);
  qkv_proj_kernel<<<gProj, 128, 0, stream>>>(Xh, Wkh, nullptr, Kh, 0);
  qkv_proj_kernel<<<gProj, 128, 0, stream>>>(Xh, Wvh, bv,      Vt, 1);

  // Pass 2: causal flash attention.
  attn_kernel<<<dim3(QBLOCKS, NBATCH * NHEADS), 128, 0, stream>>>(Qh, Kh, Vt, Cx);

  // Pass 3: output projection (f32 out).
  out_proj_kernel<<<gProj, 128, 0, stream>>>(Cx, Woh, bo, out);
}
